// CrossMolGroupInter_29326036697581
// MI455X (gfx1250) — compile-verified, matrix operand order checked by reading, not run on hardware
//
#include <hip/hip_runtime.h>
#include <cmath>

// ---------------------------------------------------------------------------
// CrossMolGroupInter forward for MI455X (gfx1250, wave32, WMMA).
// Heavy GEMMs: v_wmma_f32_16x16x32_bf16 (bf16 operands, f32 accumulate).
// QKV activations are kept in bf16 in global memory (only WMMA consumes them);
// attention staging uses global_load_async_to_lds_b128 when the toolchain
// exposes it (ASYNCcnt path), else vector load + ds_store_b128.
// Workspace requirement: ~860 MB.
// ---------------------------------------------------------------------------

#define KG     3
#define GDIM   238
#define MEDIM  18
#define EDIM   256
#define E3     768
#define NH     4
#define HD     64
#define BB     512
#define LM     192
#define NTOK   98304
#define G0SZ   40960
#define G1SZ   32768
#define G2SZ   24576

typedef __attribute__((ext_vector_type(16))) __bf16 v16bf;
typedef __attribute__((ext_vector_type(8)))  float  v8f;
typedef __attribute__((__vector_size__(16))) int    v4i32;

#if defined(__has_builtin)
#if __has_builtin(__builtin_amdgcn_global_load_async_to_lds_b128) && \
    __has_builtin(__builtin_amdgcn_s_wait_asynccnt)
#define HAVE_ASYNC 1
#endif
#endif
#ifndef HAVE_ASYNC
#define HAVE_ASYNC 0
#endif

// ---- helpers ---------------------------------------------------------------

__device__ __forceinline__ __bf16 f2bf(float f) {
  union { unsigned u; float f; } in; in.f = f;
  unsigned r = (in.u + 0x7FFFu + ((in.u >> 16) & 1u)) >> 16;   // RNE
  union { unsigned short s; __bf16 b; } out; out.s = (unsigned short)r;
  return out.b;
}
__device__ __forceinline__ unsigned pack2bf(float a, float b) {
  union { unsigned u; float f; } x, y; x.f = a; y.f = b;
  unsigned ra = (x.u + 0x7FFFu + ((x.u >> 16) & 1u)) >> 16;
  unsigned rb = (y.u + 0x7FFFu + ((y.u >> 16) & 1u)) >> 16;
  return (rb << 16) | (ra & 0xFFFFu);
}

// 16B global -> LDS copy; async (ASYNCcnt) when available.
__device__ __forceinline__ void cp16_to_lds(const void* g, void* l) {
#if HAVE_ASYNC
  __builtin_amdgcn_global_load_async_to_lds_b128(
      (__attribute__((address_space(1))) v4i32*)(void*)g,
      (__attribute__((address_space(3))) v4i32*)l,
      0, 0);
#else
  *(uint4*)l = *(const uint4*)g;
#endif
}
__device__ __forceinline__ void cp_wait() {
#if HAVE_ASYNC
  __builtin_amdgcn_s_wait_asynccnt(0);
#endif
}

// Fragment loads per CDNA5 ISA 16-bit matrix layouts (CDNA5 ISA §7.12.2).
// lane = (l16, hf); element pair v,j maps to k = 16*(v>>2) + 8*hf + 2*(v&3) + j
__device__ __forceinline__ v16bf frag_nk(const __bf16* t, int ld, int r, int hf) {
  v16bf f;
#pragma unroll
  for (int v = 0; v < 8; ++v) {
    int k = 16 * (v >> 2) + 8 * hf + 2 * (v & 3);
    f[2 * v]     = t[r * ld + k];
    f[2 * v + 1] = t[r * ld + k + 1];
  }
  return f;
}
__device__ __forceinline__ v16bf frag_kn(const __bf16* t, int ld, int n, int hf) {
  v16bf f;
#pragma unroll
  for (int v = 0; v < 8; ++v) {
    int k = 16 * (v >> 2) + 8 * hf + 2 * (v & 3);
    f[2 * v]     = t[k * ld + n];
    f[2 * v + 1] = t[(k + 1) * ld + n];
  }
  return f;
}
__device__ __forceinline__ v8f wmma_bf16(v16bf a, v16bf b, v8f c) {
  return __builtin_amdgcn_wmma_f32_16x16x32_bf16(false, a, false, b, (short)0, c,
                                                 false, false);
}

__device__ __forceinline__ float sigm(float x) { return 1.f / (1.f + __expf(-x)); }

__device__ __forceinline__ int lowb(const int* a, int n, int v) {
  int lo = 0, hi = n;
  while (lo < hi) { int md = (lo + hi) >> 1; if (a[md] < v) lo = md + 1; else hi = md; }
  return lo;
}

// ---- batch metadata --------------------------------------------------------

__global__ void meta_kernel(const int* __restrict__ gb1, const int* __restrict__ gb2,
                            const int* __restrict__ gb3, int* __restrict__ meta,
                            float* __restrict__ counts_out) {
  int b = blockIdx.x * blockDim.x + threadIdx.x;
  if (b >= BB) return;
  int s1 = lowb(gb1, G0SZ, b), c1 = lowb(gb1, G0SZ, b + 1) - s1;
  int s2 = lowb(gb2, G1SZ, b), c2 = lowb(gb2, G1SZ, b + 1) - s2;
  int s3 = lowb(gb3, G2SZ, b), c3 = lowb(gb3, G2SZ, b + 1) - s3;
  meta[b * 8 + 0] = s1; meta[b * 8 + 1] = c1;
  meta[b * 8 + 2] = s2; meta[b * 8 + 3] = c2;
  meta[b * 8 + 4] = s3; meta[b * 8 + 5] = c3;
  meta[b * 8 + 6] = c1 + c2 + c3; meta[b * 8 + 7] = 0;
  counts_out[b] = (float)(c1 + c2 + c3);
}

// ---- padded token build ----------------------------------------------------

__global__ __launch_bounds__(256) void build_padded_kernel(
    const float* __restrict__ xg1, const float* __restrict__ xg2,
    const float* __restrict__ xg3, const float* __restrict__ Wme,
    const int* __restrict__ meta, float* __restrict__ padded) {
  int row = blockIdx.x;                 // b*LM + pos
  int b = row / LM, pos = row % LM;
  int c1 = meta[b * 8 + 1], c2 = meta[b * 8 + 3], c3 = meta[b * 8 + 5];
  int tid = threadIdx.x;
  float val = 0.f;
  const float* xg = nullptr; int grp = 0, jl = 0; bool ok = false;
  if (pos < c1)                { grp = 0; jl = meta[b * 8 + 0] + pos;            xg = xg1; ok = true; }
  else if (pos < c1 + c2)      { grp = 1; jl = meta[b * 8 + 2] + pos - c1;       xg = xg2; ok = true; }
  else if (pos < c1 + c2 + c3) { grp = 2; jl = meta[b * 8 + 4] + pos - c1 - c2;  xg = xg3; ok = true; }
  if (ok)
    val = (tid < GDIM) ? xg[(long)jl * GDIM + tid] : Wme[(tid - GDIM) * KG + grp];
  padded[(long)row * EDIM + tid] = val;
}

// ---- generic GEMM: Y = act( X(MxK) @ W(NxK)^T + bias [+ Y] ), bf16 WMMA ----
// flags: bit0 = relu, bit1 = accumulate into Y, bit2 = write bf16 output
// Macro-tile 128x64, BK=32; 8 waves, each wave owns a 32x32 (2x2 WMMA) tile.

__global__ __launch_bounds__(256) void gemm_bias_kernel(
    const float* __restrict__ X, const float* __restrict__ W,
    const float* __restrict__ bias, float* __restrict__ Y,
    int M, int N, int K, int flags) {
  __shared__ __bf16 Xs[128][36];
  __shared__ __bf16 Ws[64][36];
  const int tid = threadIdx.x;
  const int lane = tid & 31, wave = tid >> 5;
  const int hf = lane >> 4, l16 = lane & 15;
  const int tm = blockIdx.y * 128, tn = blockIdx.x * 64;
  const int wm = wave >> 1, wn = wave & 1;
  const bool fast = (tm + 128 <= M) && (tn + 64 <= N) && ((K & 31) == 0);
  v8f acc[2][2] = {};
  for (int kk = 0; kk < K; kk += 32) {
    if (fast) {
#pragma unroll
      for (int i = 0; i < 4; ++i) {
        int idx = tid + 256 * i;             // 1024 float4 chunks of X tile
        int r = idx >> 3, c4 = (idx & 7) * 4;
        float4 xv = *(const float4*)&X[(long)(tm + r) * K + kk + c4];
        uint2 p; p.x = pack2bf(xv.x, xv.y); p.y = pack2bf(xv.z, xv.w);
        *(uint2*)&Xs[r][c4] = p;
      }
#pragma unroll
      for (int i = 0; i < 2; ++i) {
        int idx = tid + 256 * i;             // 512 float4 chunks of W tile
        int r = idx >> 3, c4 = (idx & 7) * 4;
        float4 wv = *(const float4*)&W[(long)(tn + r) * K + kk + c4];
        uint2 p; p.x = pack2bf(wv.x, wv.y); p.y = pack2bf(wv.z, wv.w);
        *(uint2*)&Ws[r][c4] = p;
      }
      if (kk + 32 < K) {                     // gfx1250 global_prefetch_b8
        __builtin_prefetch(&X[(long)(tm + (tid >> 3)) * K + kk + 32 + (tid & 7) * 4], 0, 1);
        __builtin_prefetch(&W[(long)(tn + ((tid >> 3) & 63)) * K + kk + 32 + (tid & 7) * 4], 0, 1);
      }
    } else {
      for (int idx = tid; idx < 128 * 32; idx += 256) {
        int r = idx >> 5, cc = idx & 31;
        int gr = tm + r, gk = kk + cc;
        Xs[r][cc] = (gr < M && gk < K) ? f2bf(X[(long)gr * K + gk]) : f2bf(0.f);
      }
      for (int idx = tid; idx < 64 * 32; idx += 256) {
        int r = idx >> 5, cc = idx & 31;
        int gr = tn + r, gk = kk + cc;
        Ws[r][cc] = (gr < N && gk < K) ? f2bf(W[(long)gr * K + gk]) : f2bf(0.f);
      }
    }
    __syncthreads();
    v16bf a0 = frag_nk(&Xs[0][0], 36, wm * 32 + l16, hf);
    v16bf a1 = frag_nk(&Xs[0][0], 36, wm * 32 + 16 + l16, hf);
    v16bf b0 = frag_nk(&Ws[0][0], 36, wn * 32 + l16, hf);
    v16bf b1 = frag_nk(&Ws[0][0], 36, wn * 32 + 16 + l16, hf);
    acc[0][0] = wmma_bf16(a0, b0, acc[0][0]);
    acc[0][1] = wmma_bf16(a0, b1, acc[0][1]);
    acc[1][0] = wmma_bf16(a1, b0, acc[1][0]);
    acc[1][1] = wmma_bf16(a1, b1, acc[1][1]);
    __syncthreads();
  }
#pragma unroll
  for (int ti = 0; ti < 2; ++ti)
#pragma unroll
    for (int tj = 0; tj < 2; ++tj) {
      v8f cc = acc[ti][tj];
#pragma unroll
      for (int v = 0; v < 8; ++v) {
        int m = tm + wm * 32 + ti * 16 + v + 8 * hf;
        int n = tn + wn * 32 + tj * 16 + l16;
        if (m < M && n < N) {
          float r = cc[v] + bias[n] + ((flags & 2) ? Y[(long)m * N + n] : 0.f);
          if (flags & 1) r = fmaxf(r, 0.f);
          if (flags & 4) ((__bf16*)Y)[(long)m * N + n] = f2bf(r);
          else           Y[(long)m * N + n] = r;
        }
      }
    }
}

// ---- attention scores: S[b,h] = Q K^T / 8 (bf16 QKV in global) ------------

__global__ __launch_bounds__(256) void scores_kernel(const __bf16* __restrict__ qkvb,
                                                     const int* __restrict__ meta,
                                                     float* __restrict__ S) {
  __shared__ __bf16 qs[LM][72];
  __shared__ __bf16 ks[LM][72];
  const int bh = blockIdx.x, b = bh >> 2, h = bh & 3;
  const int tid = threadIdx.x, lane = tid & 31, wave = tid >> 5;
  const int hf = lane >> 4, l16 = lane & 15;
  const int cnt = meta[b * 8 + 6];
  for (int ch = tid; ch < LM * 8; ch += 256) {     // 16B chunks (8 bf16)
    int r = ch >> 3, cq = (ch & 7) * 8;
    const __bf16* src = qkvb + ((long)(b * LM + r)) * E3 + h * HD + cq;
    cp16_to_lds(src, &qs[r][cq]);
    cp16_to_lds(src + EDIM, &ks[r][cq]);
  }
  cp_wait();
  __syncthreads();
  for (int t = wave; t < 144; t += 8) {
    int mt = t / 12, nt = t % 12;
    v8f c = {};
#pragma unroll
    for (int kq = 0; kq < 2; ++kq) {
      v16bf a   = frag_nk(&qs[0][kq * 32], 72, mt * 16 + l16, hf);
      v16bf bfr = frag_nk(&ks[0][kq * 32], 72, nt * 16 + l16, hf);
      c = wmma_bf16(a, bfr, c);
    }
#pragma unroll
    for (int v = 0; v < 8; ++v) {
      int i = mt * 16 + v + 8 * hf;
      int j = nt * 16 + l16;
      float val = c[v] * 0.125f;                   // 1/sqrt(64)
      if (j >= cnt) val = -INFINITY;
      S[((long)bh * LM + i) * LM + j] = val;
    }
  }
}

// ---- row softmax over keys (one wave per row) ------------------------------

__global__ __launch_bounds__(256) void softmax_kernel(float* __restrict__ S) {
  const int tid = threadIdx.x, lane = tid & 31, wave = tid >> 5;
  long row = (long)blockIdx.x * 8 + wave;
  float* p = S + row * LM;
  float vals[6];
  float m = -INFINITY;
#pragma unroll
  for (int i = 0; i < 6; ++i) { vals[i] = p[lane + 32 * i]; m = fmaxf(m, vals[i]); }
#pragma unroll
  for (int o = 16; o; o >>= 1) m = fmaxf(m, __shfl_xor(m, o, 32));
  float s = 0.f;
#pragma unroll
  for (int i = 0; i < 6; ++i) { vals[i] = __expf(vals[i] - m); s += vals[i]; }
#pragma unroll
  for (int o = 16; o; o >>= 1) s += __shfl_xor(s, o, 32);
  float inv = 1.f / s;
#pragma unroll
  for (int i = 0; i < 6; ++i) p[lane + 32 * i] = vals[i] * inv;
}

// ---- head-mean of attention weights -> w1 / w2 -----------------------------

__global__ void mean_heads_kernel(const float* __restrict__ S, float* __restrict__ out) {
  long idx = (long)blockIdx.x * 256 + threadIdx.x;       // < 512*192*192
  long b = idx / ((long)LM * LM);
  long r = idx % ((long)LM * LM);
  const long hs = (long)LM * LM;
  out[idx] = 0.25f * (S[(b * 4 + 0) * hs + r] + S[(b * 4 + 1) * hs + r] +
                      S[(b * 4 + 2) * hs + r] + S[(b * 4 + 3) * hs + r]);
}

// ---- O[b,h] = softmax(S) @ V ----------------------------------------------

__global__ __launch_bounds__(256) void attn_o_kernel(const float* __restrict__ S,
                                                     const __bf16* __restrict__ qkvb,
                                                     float* __restrict__ O) {
  __shared__ __bf16 Ss[LM][36];
  __shared__ __bf16 Vs[32][72];
  const int bh = blockIdx.x, b = bh >> 2, h = bh & 3;
  const int tid = threadIdx.x, lane = tid & 31, wave = tid >> 5;
  const int hf = lane >> 4, l16 = lane & 15;
  const long sbase = (long)bh * LM * LM;
  v8f acc[6] = {};
  for (int kk = 0; kk < LM; kk += 32) {
#pragma unroll
    for (int i = 0; i < 6; ++i) {                // S f32 -> bf16, vectorized
      int idx = tid + 256 * i;                   // 1536 float4 chunks
      int r = idx >> 3, c4 = (idx & 7) * 4;
      float4 sv = *(const float4*)&S[sbase + (long)r * LM + kk + c4];
      uint2 p; p.x = pack2bf(sv.x, sv.y); p.y = pack2bf(sv.z, sv.w);
      *(uint2*)&Ss[r][c4] = p;
    }
    {                                            // V tile: raw bf16 16B chunks
      int r = tid >> 3, cq = (tid & 7) * 8;
      cp16_to_lds(qkvb + ((long)(b * LM + kk + r)) * E3 + 2 * EDIM + h * HD + cq,
                  &Vs[r][cq]);
    }
    cp_wait();
    __syncthreads();
#pragma unroll
    for (int t = 0; t < 6; ++t) {
      int tile = wave + t * 8;                   // 48 tiles: 12 (m) x 4 (n)
      int mt = tile >> 2, nt = tile & 3;
      v16bf a   = frag_nk(&Ss[0][0], 36, mt * 16 + l16, hf);
      v16bf bfr = frag_kn(&Vs[0][0], 72, nt * 16 + l16, hf);
      acc[t] = wmma_bf16(a, bfr, acc[t]);
    }
    __syncthreads();
  }
#pragma unroll
  for (int t = 0; t < 6; ++t) {
    int tile = wave + t * 8;
    int mt = tile >> 2, nt = tile & 3;
#pragma unroll
    for (int v = 0; v < 8; ++v) {
      int i = mt * 16 + v + 8 * hf;
      int d = nt * 16 + l16;
      O[((long)(b * LM + i)) * EDIM + h * HD + d] = acc[t][v];
    }
  }
}

// ---- residual + LayerNorm (in-place on padded) -----------------------------

__global__ __launch_bounds__(256) void add_ln_kernel(float* __restrict__ x,
                                                     const float* __restrict__ a,
                                                     const float* __restrict__ g,
                                                     const float* __restrict__ bta) {
  __shared__ float red[256];
  long row = blockIdx.x;
  int tid = threadIdx.x;
  float t = x[row * EDIM + tid] + a[row * EDIM + tid];
  red[tid] = t; __syncthreads();
  for (int s = 128; s; s >>= 1) { if (tid < s) red[tid] += red[tid + s]; __syncthreads(); }
  float mean = red[0] * (1.f / EDIM); __syncthreads();
  float d = t - mean;
  red[tid] = d * d; __syncthreads();
  for (int s = 128; s; s >>= 1) { if (tid < s) red[tid] += red[tid + s]; __syncthreads(); }
  float var = red[0] * (1.f / EDIM);
  x[row * EDIM + tid] = d * rsqrtf(var + 1e-5f) * g[tid] + bta[tid];
}

// ---- unsort back to original concat order ----------------------------------

__global__ __launch_bounds__(256) void unsort_kernel(
    const float* __restrict__ padded, const int* __restrict__ gb1,
    const int* __restrict__ gb2, const int* __restrict__ gb3,
    const int* __restrict__ meta, float* __restrict__ xu) {
  int j = blockIdx.x;
  int grp, jl; const int* gb;
  if (j < G0SZ)             { grp = 0; jl = j;               gb = gb1; }
  else if (j < G0SZ + G1SZ) { grp = 1; jl = j - G0SZ;        gb = gb2; }
  else                      { grp = 2; jl = j - G0SZ - G1SZ; gb = gb3; }
  int b = gb[jl];
  int s = meta[b * 8 + 2 * grp];
  int prefix = (grp > 0 ? meta[b * 8 + 1] : 0) + (grp > 1 ? meta[b * 8 + 3] : 0);
  int pos = prefix + (jl - s);
  int tid = threadIdx.x;
  xu[(long)j * EDIM + tid] = padded[((long)(b * LM + pos)) * EDIM + tid];
}

// ---- small utilities -------------------------------------------------------

__global__ void zero_kernel(float* p, long n) {
  for (long i = (long)blockIdx.x * blockDim.x + threadIdx.x; i < n;
       i += (long)gridDim.x * blockDim.x) p[i] = 0.f;
}
__global__ void copy_kernel(const float* __restrict__ s, float* __restrict__ d, long n) {
  for (long i = (long)blockIdx.x * blockDim.x + threadIdx.x; i < n;
       i += (long)gridDim.x * blockDim.x) d[i] = s[i];
}

// ---- Set2Set LSTM cell elementwise -----------------------------------------

__global__ __launch_bounds__(256) void lstm_elt_kernel(const float* __restrict__ gates,
                                                       float* __restrict__ h,
                                                       float* __restrict__ c,
                                                       float* __restrict__ q_star) {
  int b = blockIdx.x, e = threadIdx.x;
  const float* g = gates + (long)b * 1024;
  float gi = g[e], gf = g[256 + e], gg = g[512 + e], go = g[768 + e];
  float cc = c[b * 256 + e];
  cc = sigm(gf) * cc + sigm(gi) * tanhf(gg);
  float hh = sigm(go) * tanhf(cc);
  c[b * 256 + e] = cc;
  h[b * 256 + e] = hh;
  q_star[b * 512 + e] = hh;               // q part of q_star = [q, r]
}

// ---- Set2Set segment softmax attention (runs contiguous per group) ---------

__global__ __launch_bounds__(256) void s2s_attend_kernel(
    const float* __restrict__ x, const float* __restrict__ h,
    float* __restrict__ q_star, const int* __restrict__ meta, int gmask) {
  __shared__ float hq[256];
  __shared__ float e_s[LM];
  __shared__ float a_s[LM];
  __shared__ int   row_s[LM];
  __shared__ float red[256];
  int b = blockIdx.x, tid = threadIdx.x, lane = tid & 31, wave = tid >> 5;
  hq[tid] = h[b * 256 + tid];
  int rb[3], rc[3], nr = 0, tot = 0;
  const int gbase[3] = {0, G0SZ, G0SZ + G1SZ};
#pragma unroll
  for (int i = 0; i < 3; ++i)
    if ((gmask >> i) & 1) {
      rb[nr] = gbase[i] + meta[b * 8 + 2 * i];
      rc[nr] = meta[b * 8 + 2 * i + 1];
      tot += rc[nr]; nr++;
    }
  if (tot > LM) tot = LM;
  __syncthreads();
  for (int ln = wave; ln < tot; ln += 8) {
    int t = ln, row = rb[0];
    for (int r = 0; r < nr; ++r) { if (t < rc[r]) { row = rb[r] + t; break; } t -= rc[r]; }
    float p = 0.f;
#pragma unroll
    for (int q = 0; q < 8; ++q) { int col = lane + 32 * q; p += x[(long)row * EDIM + col] * hq[col]; }
#pragma unroll
    for (int o = 16; o; o >>= 1) p += __shfl_xor(p, o, 32);
    if (lane == 0) { e_s[ln] = p; row_s[ln] = row; }
  }
  __syncthreads();
  red[tid] = (tid < tot) ? e_s[tid] : -INFINITY; __syncthreads();
  for (int s = 128; s; s >>= 1) { if (tid < s) red[tid] = fmaxf(red[tid], red[tid + s]); __syncthreads(); }
  float m = red[0]; __syncthreads();
  float av = (tid < tot) ? __expf(e_s[tid] - m) : 0.f;
  red[tid] = av; __syncthreads();
  for (int s = 128; s; s >>= 1) { if (tid < s) red[tid] += red[tid + s]; __syncthreads(); }
  float ssum = red[0];
  if (tid < tot) a_s[tid] = av / ssum;
  __syncthreads();
  float acc = 0.f;
  for (int ln = 0; ln < tot; ++ln) acc += a_s[ln] * x[(long)row_s[ln] * EDIM + tid];
  q_star[b * 512 + 256 + tid] = acc;      // r part of q_star
}

// ---------------------------------------------------------------------------

extern "C" void kernel_launch(void* const* d_in, const int* in_sizes, int n_in,
                              void* d_out, int out_size, void* d_ws, size_t ws_size,
                              hipStream_t stream) {
  (void)in_sizes; (void)n_in; (void)out_size; (void)ws_size;

  const float* xg1 = (const float*)d_in[0];
  const int*   gb1 = (const int*)d_in[1];
  const float* xg2 = (const float*)d_in[2];
  const int*   gb2 = (const int*)d_in[3];
  const float* xg3 = (const float*)d_in[4];
  const int*   gb3 = (const int*)d_in[5];
  const float* Wme = (const float*)d_in[6];
  const float* Wi[2]   = {(const float*)d_in[7],  (const float*)d_in[11]};
  const float* bi[2]   = {(const float*)d_in[8],  (const float*)d_in[12]};
  const float* Wo[2]   = {(const float*)d_in[9],  (const float*)d_in[13]};
  const float* bo[2]   = {(const float*)d_in[10], (const float*)d_in[14]};
  const float* ln_g[2] = {(const float*)d_in[15], (const float*)d_in[17]};
  const float* ln_b[2] = {(const float*)d_in[16], (const float*)d_in[18]};
  const float* R1  = (const float*)d_in[19];
  const float* rb1 = (const float*)d_in[20];
  const float* R2  = (const float*)d_in[21];
  const float* rb2 = (const float*)d_in[22];
  const float* Wih_m = (const float*)d_in[23];
  const float* Whh_m = (const float*)d_in[24];
  const float* bih_m = (const float*)d_in[25];
  const float* bhh_m = (const float*)d_in[26];
  const float* Wih_x = (const float*)d_in[27];
  const float* Whh_x = (const float*)d_in[28];
  const float* bih_x = (const float*)d_in[29];
  const float* bhh_x = (const float*)d_in[30];

  float* out = (float*)d_out;
  // output layout: per_mol(3*512*238) | mix(512*512) | w1 | w2 | counts(512)
  const long PM_OFF  = 0;
  const long MIX_OFF = 365568;
  const long W1_OFF  = 627712;
  const long W2_OFF  = W1_OFF + (long)BB * LM * LM;     // 19502080
  const long CNT_OFF = W2_OFF + (long)BB * LM * LM;     // 38376448

  char* cur = (char*)d_ws;
  auto alloc = [&](size_t bytes) -> void* {
    void* p = cur; cur += (bytes + 255) & ~(size_t)255; return p;
  };
  float*  padded = (float*)alloc((size_t)NTOK * EDIM * 4);
  __bf16* qkvb   = (__bf16*)alloc((size_t)NTOK * E3 * 2);
  float*  S      = (float*)alloc((size_t)BB * NH * LM * LM * 4);
  float*  attnO  = (float*)alloc((size_t)NTOK * EDIM * 4);
  float*  proj   = (float*)alloc((size_t)NTOK * EDIM * 4);
  float*  xu     = (float*)alloc((size_t)NTOK * EDIM * 4);
  float*  qstar  = (float*)alloc((size_t)BB * 512 * 4);
  float*  hbuf   = (float*)alloc((size_t)BB * 256 * 4);
  float*  cbuf   = (float*)alloc((size_t)BB * 256 * 4);
  float*  gates  = (float*)alloc((size_t)BB * 1024 * 4);
  float*  tmp1   = (float*)alloc((size_t)BB * GDIM * 4);
  int*    meta   = (int*)alloc((size_t)BB * 8 * 4);

  auto gemm = [&](const float* X, const float* W, const float* bias, float* Y,
                  int M, int N, int Kd, int flags) {
    dim3 grid((N + 63) / 64, (M + 127) / 128);
    gemm_bias_kernel<<<grid, 256, 0, stream>>>(X, W, bias, Y, M, N, Kd, flags);
  };

  meta_kernel<<<2, 256, 0, stream>>>(gb1, gb2, gb3, meta, out + CNT_OFF);
  build_padded_kernel<<<NTOK, 256, 0, stream>>>(xg1, xg2, xg3, Wme, meta, padded);

  for (int l = 0; l < 2; ++l) {
    gemm(padded, Wi[l], bi[l], (float*)qkvb, NTOK, E3, EDIM, 4 /*bf16 out*/);
    scores_kernel<<<BB * NH, 256, 0, stream>>>(qkvb, meta, S);
    softmax_kernel<<<BB * NH * LM / 8, 256, 0, stream>>>(S);
    mean_heads_kernel<<<(int)(((long)BB * LM * LM) / 256), 256, 0, stream>>>(
        S, out + (l == 0 ? W1_OFF : W2_OFF));
    attn_o_kernel<<<BB * NH, 256, 0, stream>>>(S, qkvb, attnO);
    gemm(attnO, Wo[l], bo[l], proj, NTOK, EDIM, EDIM, 0);
    add_ln_kernel<<<NTOK, 256, 0, stream>>>(padded, proj, ln_g[l], ln_b[l]);
  }

  unsort_kernel<<<NTOK, 256, 0, stream>>>(padded, gb1, gb2, gb3, meta, xu);

  auto s2s = [&](const float* Wih, const float* Whh, const float* bih,
                 const float* bhh, int gmask) {
    zero_kernel<<<256, 256, 0, stream>>>(qstar, (long)BB * 512);
    zero_kernel<<<256, 256, 0, stream>>>(hbuf, (long)BB * 256);
    zero_kernel<<<256, 256, 0, stream>>>(cbuf, (long)BB * 256);
    for (int s = 0; s < 2; ++s) {
      gemm(qstar, Wih, bih, gates, BB, 1024, 512, 0);
      gemm(hbuf,  Whh, bhh, gates, BB, 1024, 256, 2);  // accumulate
      lstm_elt_kernel<<<BB, 256, 0, stream>>>(gates, hbuf, cbuf, qstar);
      s2s_attend_kernel<<<BB, 256, 0, stream>>>(xu, hbuf, qstar, meta, gmask);
    }
  };

  for (int g = 0; g < 3; ++g) {
    s2s(Wih_m, Whh_m, bih_m, bhh_m, 1 << g);
    gemm(qstar, R1, rb1, tmp1, BB, GDIM, 512, 1 /*relu*/);
    gemm(tmp1, R2, rb2, out + PM_OFF + (long)g * BB * GDIM, BB, GDIM, GDIM, 0);
  }
  s2s(Wih_x, Whh_x, bih_x, bhh_x, 7);
  copy_kernel<<<256, 256, 0, stream>>>(qstar, out + MIX_OFF, (long)BB * 512);
}
